// CausalSelfAttention_4947802325800
// MI455X (gfx1250) — matrix operand order repaired and checked by reading, not compiled, not run on hardware
//
#include <hip/hip_runtime.h>

// ---------------------------------------------------------------------------
// Causal self-attention: B=4, T=2048, D=1024, H=16, HD=64
// All GEMMs on v_wmma_f32_16x16x32_bf16 (wave32, gfx1250).
// ---------------------------------------------------------------------------

#define B_  4
#define T_  2048
#define D_  1024
#define H_  16
#define HD_ 64

typedef __attribute__((ext_vector_type(16))) __bf16 v16bf;
typedef __attribute__((ext_vector_type(8)))  __bf16 v8bf;
typedef __attribute__((ext_vector_type(8)))  float  v8f;

union Frag { v16bf v; v8bf h[2]; };

static __device__ __forceinline__ __bf16 f2bf(float f) {
  unsigned int u = __builtin_bit_cast(unsigned int, f);
  u += 0x7fffu + ((u >> 16) & 1u);          // round-to-nearest-even
  unsigned short s = (unsigned short)(u >> 16);
  return __builtin_bit_cast(__bf16, s);
}

static __device__ __forceinline__ v8bf ldg8(const __bf16* p) {
  return *(const v8bf*)p;                    // 16B -> global_load_b128
}

static __device__ __forceinline__ v8f wmma_bf16(const Frag& a, const Frag& b, v8f c) {
  return __builtin_amdgcn_wmma_f32_16x16x32_bf16(
      /*neg_a=*/false, a.v, /*neg_b=*/false, b.v,
      /*c_mod=*/(short)0, c, /*reuse_a=*/false, /*reuse_b=*/false);
}

// --------------------------- conversion kernels ----------------------------

__global__ __launch_bounds__(256) void k_cvt(const float* __restrict__ in,
                                             __bf16* __restrict__ out, int n) {
  int i = blockIdx.x * 256 + threadIdx.x;
  if (i < n) out[i] = f2bf(in[i]);
}

// in: [rows][cols] row-major fp32  ->  out: [cols][rows] bf16 (transposed)
__global__ __launch_bounds__(256) void k_cvt_tr(const float* __restrict__ in,
                                                __bf16* __restrict__ out,
                                                int rows, int cols) {
  int i = blockIdx.x * 256 + threadIdx.x;
  if (i < rows * cols) {
    int r = i / cols, c = i % cols;
    out[(size_t)c * rows + r] = f2bf(in[i]);
  }
}

// ------------------------------ QKV GEMM -----------------------------------
// xb:[B*T][D] bf16, wT:[3D][D] bf16 (transposed).  Per wave: 32x64 tile.
// Writes q,k as [B,H,T,HD] and v transposed as [B,H,HD,T].

__global__ __launch_bounds__(128) void k_gemm_qkv(
    const __bf16* __restrict__ xb, const __bf16* __restrict__ wT,
    __bf16* __restrict__ qo, __bf16* __restrict__ ko, __bf16* __restrict__ vto) {
  const int K    = D_;
  const int lane = threadIdx.x & 31;
  const int wid  = threadIdx.x >> 5;
  const int wtile = blockIdx.x * 4 + wid;      // 256*48 = 12288 tiles
  const int nt = wtile % 48;
  const int mt = wtile / 48;
  const int m0 = mt * 32, n0 = nt * 64;
  const int lr = lane & 15, lh = lane >> 4;

  const v8f z8 = {0.f,0.f,0.f,0.f,0.f,0.f,0.f,0.f};
  v8f acc[2][4];
  #pragma unroll
  for (int im = 0; im < 2; ++im)
    #pragma unroll
    for (int jn = 0; jn < 4; ++jn) acc[im][jn] = z8;

  for (int k0 = 0; k0 < K; k0 += 32) {
    Frag a[2];
    #pragma unroll
    for (int im = 0; im < 2; ++im) {
      const __bf16* p = xb + (size_t)(m0 + im*16 + lr) * K + k0 + lh*8;
      a[im].h[0] = ldg8(p); a[im].h[1] = ldg8(p + 16);
    }
    if (k0 + 64 < K) {
      __builtin_prefetch(xb + (size_t)(m0 + lr) * K + k0 + 64, 0, 1);
      __builtin_prefetch(wT + (size_t)(n0 + lr) * K + k0 + 64, 0, 1);
    }
    #pragma unroll
    for (int jn = 0; jn < 4; ++jn) {
      Frag bfr;
      const __bf16* p = wT + (size_t)(n0 + jn*16 + lr) * K + k0 + lh*8;
      bfr.h[0] = ldg8(p); bfr.h[1] = ldg8(p + 16);
      acc[0][jn] = wmma_bf16(a[0], bfr, acc[0][jn]);
      acc[1][jn] = wmma_bf16(a[1], bfr, acc[1][jn]);
    }
  }

  #pragma unroll
  for (int im = 0; im < 2; ++im)
    #pragma unroll
    for (int jn = 0; jn < 4; ++jn)
      #pragma unroll
      for (int i = 0; i < 8; ++i) {
        int m = m0 + im*16 + i + 8*lh;          // C-layout row
        int n = n0 + jn*16 + lr;                // C-layout col
        __bf16 val = f2bf(acc[im][jn][i]);
        int which = n >> 10, c = n & 1023;
        int hh = c >> 6, hd = c & 63;
        int bb = m >> 11, t = m & 2047;
        if (which == 0)
          qo[(((size_t)(bb*H_ + hh))*T_ + t)*HD_ + hd] = val;
        else if (which == 1)
          ko[(((size_t)(bb*H_ + hh))*T_ + t)*HD_ + hd] = val;
        else
          vto[(((size_t)(bb*H_ + hh))*HD_ + hd)*T_ + t] = val;
      }
}

// --------------------------- flash attention -------------------------------
// One wave = 16 query rows; stream 32-key blocks with online softmax.
// q,k: [B,H,T,HD]  vt: [B,H,HD,T]  y: [B,T,D] bf16.

__global__ __launch_bounds__(128) void k_attn(
    const __bf16* __restrict__ qg, const __bf16* __restrict__ kg,
    const __bf16* __restrict__ vtg, __bf16* __restrict__ yg) {
  __shared__ __bf16 plds[4 * 16 * 32];         // per-wave 16x32 P tile
  const int lane = threadIdx.x & 31, wid = threadIdx.x >> 5;
  const int nqb = T_ / 64;
  int blk = blockIdx.x;
  int qb = blk % nqb;
  int hh = (blk / nqb) % H_;
  int bb = blk / (nqb * H_);
  const __bf16* Q  = qg  + ((size_t)(bb*H_ + hh) * T_) * HD_;
  const __bf16* Kp = kg  + ((size_t)(bb*H_ + hh) * T_) * HD_;
  const __bf16* Vt = vtg + ((size_t)(bb*H_ + hh) * HD_) * T_;
  const int q0 = qb*64 + wid*16;
  const int lr = lane & 15, lh = lane >> 4;

  // Q A-fragments, reused for the whole key loop (HD=64 -> two K=32 chunks)
  Frag aq[2];
  #pragma unroll
  for (int kc = 0; kc < 2; ++kc) {
    const __bf16* p = Q + (size_t)(q0 + lr)*HD_ + kc*32 + lh*8;
    aq[kc].h[0] = ldg8(p); aq[kc].h[1] = ldg8(p + 16);
  }

  const v8f z8 = {0.f,0.f,0.f,0.f,0.f,0.f,0.f,0.f};
  v8f O[4];
  #pragma unroll
  for (int j = 0; j < 4; ++j) O[j] = z8;
  float mrow[8], lrow[8];
  #pragma unroll
  for (int i = 0; i < 8; ++i) { mrow[i] = -1e30f; lrow[i] = 0.f; }

  __bf16* pl = plds + wid * 512;
  const int kb_end = (q0 + 16 + 31) >> 5;
  const float scale = 0.125f;                   // 1/sqrt(64)

  for (int kb = 0; kb < kb_end; ++kb) {
    v8f S[2]; S[0] = z8; S[1] = z8;
    #pragma unroll
    for (int t = 0; t < 2; ++t)
      #pragma unroll
      for (int kc = 0; kc < 2; ++kc) {
        Frag bk;
        const __bf16* p = Kp + (size_t)(kb*32 + t*16 + lr)*HD_ + kc*32 + lh*8;
        bk.h[0] = ldg8(p); bk.h[1] = ldg8(p + 16);
        S[t] = wmma_bf16(aq[kc], bk, S[t]);
      }

    // scale + causal mask + row max (rows for this lane: q0 + 8*lh + i)
    const int rowa = q0 + 8*lh;
    const int key0 = kb*32 + lr, key1 = key0 + 16;
    float bm[8];
    #pragma unroll
    for (int i = 0; i < 8; ++i) {
      float s0 = S[0][i] * scale, s1 = S[1][i] * scale;
      int r = rowa + i;
      s0 = (key0 > r) ? -1e30f : s0;
      s1 = (key1 > r) ? -1e30f : s1;
      S[0][i] = s0; S[1][i] = s1;
      bm[i] = fmaxf(s0, s1);
    }
    #pragma unroll
    for (int off = 8; off >= 1; off >>= 1)
      #pragma unroll
      for (int i = 0; i < 8; ++i)
        bm[i] = fmaxf(bm[i], __shfl_xor(bm[i], off, 32));

    float alpha[8], rs[8];
    #pragma unroll
    for (int i = 0; i < 8; ++i) {
      float mn = fmaxf(mrow[i], bm[i]);
      alpha[i] = __expf(mrow[i] - mn);
      mrow[i] = mn;
      float p0 = __expf(S[0][i] - mn);
      float p1 = __expf(S[1][i] - mn);
      S[0][i] = p0; S[1][i] = p1;
      rs[i] = p0 + p1;
    }
    #pragma unroll
    for (int off = 8; off >= 1; off >>= 1)
      #pragma unroll
      for (int i = 0; i < 8; ++i)
        rs[i] += __shfl_xor(rs[i], off, 32);
    #pragma unroll
    for (int i = 0; i < 8; ++i) lrow[i] = lrow[i]*alpha[i] + rs[i];
    #pragma unroll
    for (int j = 0; j < 4; ++j)
      #pragma unroll
      for (int i = 0; i < 8; ++i) O[j][i] *= alpha[i];

    // P: C-layout -> LDS -> A-layout (same-wave LDS is in-order)
    #pragma unroll
    for (int i = 0; i < 8; ++i) {
      pl[(i + 8*lh)*32 + lr]      = f2bf(S[0][i]);
      pl[(i + 8*lh)*32 + 16 + lr] = f2bf(S[1][i]);
    }
    asm volatile("s_wait_dscnt 0x0" ::: "memory");
    Frag ap;
    {
      const __bf16* p = pl + lr*32 + lh*8;
      ap.h[0] = *(const v8bf*)p;
      ap.h[1] = *(const v8bf*)(p + 16);
    }
    // O += P @ V  (B-fragment columns contiguous thanks to V-transpose)
    #pragma unroll
    for (int j = 0; j < 4; ++j) {
      Frag bv;
      const __bf16* p = Vt + (size_t)(j*16 + lr)*T_ + kb*32 + lh*8;
      bv.h[0] = ldg8(p); bv.h[1] = ldg8(p + 16);
      O[j] = wmma_bf16(ap, bv, O[j]);
    }
    asm volatile("" ::: "memory");   // keep next-iter LDS stores after reads
  }

  #pragma unroll
  for (int i = 0; i < 8; ++i) lrow[i] = 1.0f / lrow[i];
  #pragma unroll
  for (int j = 0; j < 4; ++j)
    #pragma unroll
    for (int i = 0; i < 8; ++i) {
      int t   = q0 + i + 8*lh;
      int col = hh*HD_ + j*16 + lr;
      yg[((size_t)bb*T_ + t)*D_ + col] = f2bf(O[j][i] * lrow[i]);
    }
}

// ---------------------------- output GEMM ----------------------------------
// yb:[B*T][D] bf16, wT:[D][D] bf16 (transposed), out:[B*T][D] fp32.

__global__ __launch_bounds__(128) void k_gemm_out(
    const __bf16* __restrict__ yb, const __bf16* __restrict__ wT,
    float* __restrict__ out) {
  const int K    = D_;
  const int lane = threadIdx.x & 31;
  const int wid  = threadIdx.x >> 5;
  const int wtile = blockIdx.x * 4 + wid;      // 256*16 = 4096 tiles
  const int nt = wtile % 16;
  const int mt = wtile / 16;
  const int m0 = mt * 32, n0 = nt * 64;
  const int lr = lane & 15, lh = lane >> 4;

  const v8f z8 = {0.f,0.f,0.f,0.f,0.f,0.f,0.f,0.f};
  v8f acc[2][4];
  #pragma unroll
  for (int im = 0; im < 2; ++im)
    #pragma unroll
    for (int jn = 0; jn < 4; ++jn) acc[im][jn] = z8;

  for (int k0 = 0; k0 < K; k0 += 32) {
    Frag a[2];
    #pragma unroll
    for (int im = 0; im < 2; ++im) {
      const __bf16* p = yb + (size_t)(m0 + im*16 + lr) * K + k0 + lh*8;
      a[im].h[0] = ldg8(p); a[im].h[1] = ldg8(p + 16);
    }
    if (k0 + 64 < K) {
      __builtin_prefetch(yb + (size_t)(m0 + lr) * K + k0 + 64, 0, 1);
      __builtin_prefetch(wT + (size_t)(n0 + lr) * K + k0 + 64, 0, 1);
    }
    #pragma unroll
    for (int jn = 0; jn < 4; ++jn) {
      Frag bfr;
      const __bf16* p = wT + (size_t)(n0 + jn*16 + lr) * K + k0 + lh*8;
      bfr.h[0] = ldg8(p); bfr.h[1] = ldg8(p + 16);
      acc[0][jn] = wmma_bf16(a[0], bfr, acc[0][jn]);
      acc[1][jn] = wmma_bf16(a[1], bfr, acc[1][jn]);
    }
  }

  #pragma unroll
  for (int im = 0; im < 2; ++im)
    #pragma unroll
    for (int jn = 0; jn < 4; ++jn)
      #pragma unroll
      for (int i = 0; i < 8; ++i) {
        int m = m0 + im*16 + i + 8*lh;
        int n = n0 + jn*16 + lr;
        out[(size_t)m * D_ + n] = acc[im][jn][i];
      }
}

// ------------------------------ launcher -----------------------------------

extern "C" void kernel_launch(void* const* d_in, const int* in_sizes, int n_in,
                              void* d_out, int out_size, void* d_ws, size_t ws_size,
                              hipStream_t stream) {
  (void)in_sizes; (void)n_in; (void)out_size; (void)ws_size;
  const float* x    = (const float*)d_in[0];
  const float* wqkv = (const float*)d_in[1];
  const float* wout = (const float*)d_in[2];
  float* out = (float*)d_out;

  char* ws = (char*)d_ws;
  size_t o = 0;
  __bf16* xb    = (__bf16*)(ws + o); o += (size_t)B_*T_*D_*2;   // x bf16
  __bf16* wqkvT = (__bf16*)(ws + o); o += (size_t)3*D_*D_*2;    // [3D][D]
  __bf16* woutT = (__bf16*)(ws + o); o += (size_t)D_*D_*2;      // [D][D]
  __bf16* qb    = (__bf16*)(ws + o); o += (size_t)B_*T_*D_*2;   // [B,H,T,HD]
  __bf16* kb    = (__bf16*)(ws + o); o += (size_t)B_*T_*D_*2;   // [B,H,T,HD]
  __bf16* vtb   = (__bf16*)(ws + o); o += (size_t)B_*T_*D_*2;   // [B,H,HD,T]
  __bf16* yb    = (__bf16*)(ws + o); o += (size_t)B_*T_*D_*2;   // [B,T,D]

  const int nx = B_*T_*D_;
  k_cvt<<<dim3((nx + 255) / 256), dim3(256), 0, stream>>>(x, xb, nx);
  k_cvt_tr<<<dim3((D_*3*D_ + 255) / 256), dim3(256), 0, stream>>>(wqkv, wqkvT, D_, 3*D_);
  k_cvt_tr<<<dim3((D_*D_ + 255) / 256), dim3(256), 0, stream>>>(wout, woutT, D_, D_);

  // 8192/32 M-tiles * 3072/64 N-tiles = 12288 wave tiles / 4 waves per block
  k_gemm_qkv<<<dim3(3072), dim3(128), 0, stream>>>(xb, wqkvT, qb, kb, vtb);

  // B*H*(T/64) = 2048 blocks, 4 waves each (16 query rows per wave)
  k_attn<<<dim3(B_*H_*(T_/64)), dim3(128), 0, stream>>>(qb, kb, vtb, yb);

  // 8192/32 * 1024/64 = 4096 wave tiles / 4 per block
  k_gemm_out<<<dim3(1024), dim3(128), 0, stream>>>(yb, woutT, out);
}
